// SDMIL_58884001628800
// MI455X (gfx1250) — matrix operand order repaired and checked by reading, not compiled, not run on hardware
//
#include <hip/hip_runtime.h>
#include <math.h>

#define GN 65536   // instances
#define GC 512     // feature dim
#define GP 512     // prototypes
#define LDSB 40    // LDS row stride in bf16 elements (32 + 8 pad; 80 B keeps b128 alignment)

typedef __attribute__((ext_vector_type(16))) __bf16        v16bf;
typedef __attribute__((ext_vector_type(8)))  float         v8f;
typedef __attribute__((ext_vector_type(4)))  float         f32x4;
typedef __attribute__((ext_vector_type(4)))  unsigned int  u32x4;

union BF16Frag { v16bf v; u32x4 q[2]; };
union AccFrag  { v8f v; float f[8]; };

// Pack the bf16(hi) of two f32s into one dword: {v1[31:16], v0[31:16]}
static __device__ __forceinline__ unsigned int pack_hi2(float v0, float v1) {
    return __builtin_amdgcn_perm(__float_as_uint(v1), __float_as_uint(v0), 0x07060302u);
}
static __device__ __forceinline__ uint2 pack_hi4(f32x4 v) {
    uint2 r;
    r.x = pack_hi2(v[0], v[1]);
    r.y = pack_hi2(v[2], v[3]);
    return r;
}
// residual after truncating to bf16: v - bf16_trunc(v)  (exact in f32)
static __device__ __forceinline__ f32x4 lo_part4(f32x4 v) {
    f32x4 r;
#pragma unroll
    for (int j = 0; j < 4; ++j)
        r[j] = v[j] - __uint_as_float(__float_as_uint(v[j]) & 0xffff0000u);
    return r;
}

// ---------------------------------------------------------------------------
// Kernel 1: row squared-norms of x (N rows) and prototype (P rows).
// One wave per row; lane-coalesced float4 reads; shfl_xor reduction.
// ---------------------------------------------------------------------------
__global__ __launch_bounds__(256) void sq_kernel(const float* __restrict__ x,
                                                 const float* __restrict__ proto,
                                                 float* __restrict__ xsq,
                                                 float* __restrict__ psq) {
    const int wave = threadIdx.x >> 5;
    const int lane = threadIdx.x & 31;
    const int row  = blockIdx.x * 8 + wave;   // grid covers GN + GP rows
    const float* src;
    float* dst;
    if (row < GN) { src = x + (size_t)row * GC;          dst = xsq + row; }
    else          { int p = row - GN; src = proto + (size_t)p * GC; dst = psq + p; }
    float s = 0.0f;
#pragma unroll
    for (int chunk = 0; chunk < 4; ++chunk) {
        f32x4 v = *(const f32x4*)(src + chunk * 128 + lane * 4);
        s += v[0]*v[0] + v[1]*v[1] + v[2]*v[2] + v[3]*v[3];
    }
#pragma unroll
    for (int off = 16; off > 0; off >>= 1) s += __shfl_xor(s, off, 32);
    if (lane == 0) *dst = s;
}

// ---------------------------------------------------------------------------
// Kernel 2: WMMA GEMM (prototype x x^T) with bf16 hi+lo split (3 WMMAs per
// tile: hi*hi + lo*hi + hi*lo ~ f32 accuracy). The f32->bf16 split is done
// ONCE per element during cooperative global->LDS staging (packed bf16 in
// LDS); per-wave fragment gather is pure ds_load_b128. Fused distance
// epilogue stores UNNORMALIZED inverse distances sqrt(C)*rsqrt(d2).
// Block tile 128(M=P) x 128(N); 8 waves as 4x2; wave tile 32x64.
// ---------------------------------------------------------------------------
__global__ __launch_bounds__(256) void gemm_kernel(const float* __restrict__ x,
                                                   const float* __restrict__ proto,
                                                   const float* __restrict__ xsq,
                                                   const float* __restrict__ psq,
                                                   float* __restrict__ wout) {
    __shared__ unsigned short sAh[128 * LDSB];   // 10 KB each
    __shared__ unsigned short sAl[128 * LDSB];
    __shared__ unsigned short sBh[128 * LDSB];
    __shared__ unsigned short sBl[128 * LDSB];

    const int tid     = threadIdx.x;
    const int lane    = tid & 31;
    const int wave    = tid >> 5;
    const int m0      = blockIdx.x * 128;     // P block (grid.x = 4)
    const int n0      = blockIdx.y * 128;     // N block (grid.y = 512)
    const int wm0     = (wave >> 1) * 32;     // wave M offset in block
    const int wn0     = (wave & 1) * 64;      // wave N offset in block
    const int lr      = lane & 15;
    const int halfsel = lane >> 4;            // 0: lanes 0-15, 1: lanes 16-31

    v8f acc[2][4];
    {
        v8f z = {};
#pragma unroll
        for (int mt = 0; mt < 2; ++mt)
#pragma unroll
            for (int nt = 0; nt < 4; ++nt) acc[mt][nt] = z;
    }

    const int ldr = tid >> 3;                 // 0..31 (row within 32-row slab)
    const int ldc = (tid & 7) * 4;            // 0..28 (float4 column in k)

    for (int kc = 0; kc < GC; kc += 32) {
        __syncthreads();
        // ---- coalesced global load + one-time bf16 hi/lo split -> LDS ----
#pragma unroll
        for (int rr = 0; rr < 128; rr += 32) {
            const int row = ldr + rr;
            f32x4 av = *(const f32x4*)(proto + (size_t)(m0 + row) * GC + kc + ldc);
            f32x4 bv = *(const f32x4*)(x     + (size_t)(n0 + row) * GC + kc + ldc);
            const int o = row * LDSB + ldc;
            *(uint2*)(sAh + o) = pack_hi4(av);
            *(uint2*)(sAl + o) = pack_hi4(lo_part4(av));
            *(uint2*)(sBh + o) = pack_hi4(bv);
            *(uint2*)(sBl + o) = pack_hi4(lo_part4(bv));
        }
        __syncthreads();

        // ---- gather packed-bf16 fragments (pure ds_load_b128) ----
        BF16Frag a_hi[2], a_lo[2], b_hi[4], b_lo[4];
        // A 16x32: lanes 0-15 K {0..7,16..23}; lanes 16-31 K {8..15,24..31}
#pragma unroll
        for (int mt = 0; mt < 2; ++mt) {
            const int o = (wm0 + mt * 16 + lr) * LDSB + halfsel * 8;
            a_hi[mt].q[0] = *(const u32x4*)(sAh + o);
            a_hi[mt].q[1] = *(const u32x4*)(sAh + o + 16);
            a_lo[mt].q[0] = *(const u32x4*)(sAl + o);
            a_lo[mt].q[1] = *(const u32x4*)(sAl + o + 16);
        }
        // B 32x16 (col = lr): lanes 0-15 K {0..7,16..23}; lanes 16-31 K {8..15,24..31}
#pragma unroll
        for (int nt = 0; nt < 4; ++nt) {
            const int o = (wn0 + nt * 16 + lr) * LDSB + halfsel * 8;
            b_hi[nt].q[0] = *(const u32x4*)(sBh + o);
            b_hi[nt].q[1] = *(const u32x4*)(sBh + o + 16);
            b_lo[nt].q[0] = *(const u32x4*)(sBl + o);
            b_lo[nt].q[1] = *(const u32x4*)(sBl + o + 16);
        }

        // ---- 2x4 tiles x 3 WMMAs (hi*hi + lo*hi + hi*lo) ----
#pragma unroll
        for (int mt = 0; mt < 2; ++mt) {
#pragma unroll
            for (int nt = 0; nt < 4; ++nt) {
                acc[mt][nt] = __builtin_amdgcn_wmma_f32_16x16x32_bf16(
                    false, a_hi[mt].v, false, b_hi[nt].v, (short)0, acc[mt][nt], false, false);
                acc[mt][nt] = __builtin_amdgcn_wmma_f32_16x16x32_bf16(
                    false, a_lo[mt].v, false, b_hi[nt].v, (short)0, acc[mt][nt], false, false);
                acc[mt][nt] = __builtin_amdgcn_wmma_f32_16x16x32_bf16(
                    false, a_hi[mt].v, false, b_lo[nt].v, (short)0, acc[mt][nt], false, false);
            }
        }
    }

    // ---- epilogue: d2 = p^2 + x^2 - 2c ; store sqrt(C)*rsqrt(d2) ----
    const float SCALE = 22.627416997969522f;  // sqrt(512)
    float pr[2][8];
#pragma unroll
    for (int mt = 0; mt < 2; ++mt)
#pragma unroll
        for (int r = 0; r < 8; ++r)
            pr[mt][r] = psq[m0 + wm0 + mt * 16 + halfsel * 8 + r];

#pragma unroll
    for (int mt = 0; mt < 2; ++mt) {
#pragma unroll
        for (int nt = 0; nt < 4; ++nt) {
            AccFrag af; af.v = acc[mt][nt];
            const int n = n0 + wn0 + nt * 16 + lr;         // C/D: N = lane%16
            const float xs = xsq[n];
            const size_t base = (size_t)(m0 + wm0 + mt * 16 + halfsel * 8) * GN + n;
#pragma unroll
            for (int r = 0; r < 8; ++r) {                  // C/D: M = r (+8 for upper half)
                float d2 = fmaxf(pr[mt][r] + xs - 2.0f * af.f[r], 0.0f);
                wout[base + (size_t)r * GN] = SCALE * rsqrtf(d2);  // d2==0 -> +inf, matches 1/0
            }
        }
    }
}

// ---------------------------------------------------------------------------
// Kernel 3: deterministic row sums of unnormalized inverse distances -> 1/sum.
// ---------------------------------------------------------------------------
__global__ __launch_bounds__(256) void rowsum_kernel(const float* __restrict__ w,
                                                     float* __restrict__ invsum) {
    __shared__ float red[256];
    const float* src = w + (size_t)blockIdx.x * GN;
    float s = 0.0f;
    for (int i = threadIdx.x * 4; i < GN; i += 1024) {
        f32x4 v = *(const f32x4*)(src + i);
        s += v[0] + v[1] + v[2] + v[3];
    }
    red[threadIdx.x] = s;
    __syncthreads();
    for (int st = 128; st > 0; st >>= 1) {
        if (threadIdx.x < st) red[threadIdx.x] += red[threadIdx.x + st];
        __syncthreads();
    }
    if (threadIdx.x == 0) invsum[blockIdx.x] = 1.0f / red[0];
}

// ---------------------------------------------------------------------------
// Kernel 4: normalize weights in place (coalesced along N) and produce
// x_out[n,c] = weights[c,n] * x[n,c] via a 32x32 LDS transpose tile
// (coalesced along C).
// ---------------------------------------------------------------------------
__global__ __launch_bounds__(256) void finalize_kernel(const float* __restrict__ x,
                                                       float* __restrict__ w,
                                                       const float* __restrict__ invsum,
                                                       float* __restrict__ xout) {
    __shared__ float tw[32][33];
    const int c0 = blockIdx.x * 32;
    const int n0 = blockIdx.y * 32;
    const int j  = threadIdx.x & 31;
    const int i0 = threadIdx.x >> 5;  // 0..7
#pragma unroll
    for (int i = i0; i < 32; i += 8) {
        const int c = c0 + i;
        const size_t idx = (size_t)c * GN + (n0 + j);
        float v = w[idx] * invsum[c];
        w[idx] = v;
        tw[i][j] = v;
    }
    __syncthreads();
#pragma unroll
    for (int i = i0; i < 32; i += 8) {
        const int n = n0 + i;
        const size_t idx = (size_t)n * GC + (c0 + j);
        xout[idx] = tw[j][i] * x[idx];   // weights[c0+j, n0+i]
    }
}

// ---------------------------------------------------------------------------
extern "C" void kernel_launch(void* const* d_in, const int* in_sizes, int n_in,
                              void* d_out, int out_size, void* d_ws, size_t ws_size,
                              hipStream_t stream) {
    const float* x     = (const float*)d_in[0];   // (1, N, C) f32
    const float* proto = (const float*)d_in[1];   // (P, C) f32

    float* out  = (float*)d_out;
    float* xout = out;                            // (N, C)
    float* wout = out + (size_t)GN * GC;          // (P, N)

    float* xsq    = (float*)d_ws;                 // N floats
    float* psq    = xsq + GN;                     // P floats
    float* invsum = psq + GP;                     // P floats  (~260 KB total ws)

    sq_kernel<<<dim3((GN + GP) / 8), 256, 0, stream>>>(x, proto, xsq, psq);
    gemm_kernel<<<dim3(GP / 128, GN / 128), 256, 0, stream>>>(x, proto, xsq, psq, wout);
    rowsum_kernel<<<dim3(GP), 256, 0, stream>>>(wout, invsum);
    finalize_kernel<<<dim3(GC / 32, GN / 32), 256, 0, stream>>>(x, wout, invsum, xout);
}
